// BaGuaLLM_2362232013037
// MI455X (gfx1250) — matrix operand order repaired and checked by reading, not compiled, not run on hardware
//
#include <hip/hip_runtime.h>
#include <hip/hip_bf16.h>
#include <math.h>

// ---------------------------------------------------------------------------
// BaGuaLLM forward for MI455X (gfx1250).
// All four GEMMs run on v_wmma_f32_16x16x32_bf16 with f32 accumulation.
// Operands are pre-converted to bf16 (weights: per-layer pack kernel with
// k-pair interleave; activations: written as bf16 by producer epilogues), so
// the GEMM inner loop is pure data movement + WMMA. Tile staging uses the
// CDNA5 async global->LDS path (ASYNCcnt) when the toolchain exposes it.
// ---------------------------------------------------------------------------

typedef __attribute__((ext_vector_type(16))) __bf16 v16bf;
typedef __attribute__((ext_vector_type(8)))  float  v8f;

union AccU { v8f v; float e[8]; };

#if defined(__HIP_DEVICE_COMPILE__) && \
    __has_builtin(__builtin_amdgcn_global_load_async_to_lds_b32) && \
    __has_builtin(__builtin_amdgcn_s_wait_asynccnt)
#define HAVE_ASYNC_LDS 1
typedef __attribute__((address_space(1))) int gint_t;   // global int*
typedef __attribute__((address_space(3))) int lint_t;   // LDS int*
#else
#define HAVE_ASYNC_LDS 0
#endif

static __device__ __forceinline__ unsigned bf16bits(float x) {
    // round-to-nearest-even f32 -> bf16 (low 16 bits of result)
    unsigned u = __builtin_bit_cast(unsigned, x);
    return ((u + 0x7FFFu + ((u >> 16) & 1u)) >> 16) & 0xFFFFu;
}

static __device__ __forceinline__ unsigned short bf16h(float x) {
    return (unsigned short)bf16bits(x);
}

static __device__ __forceinline__ float gelu_exact(float x) {
    return 0.5f * x * (1.0f + erff(x * 0.70710678118654752f));
}

// ---------------------------------------------------------------------------
// Tiled bf16 WMMA GEMM:  C[M,N] = epilogue( A[M,K] @ B[K,N] )
//   A : row-major bf16 (ushort), M x K
//   Bp: k-pair packed bf16, u32[(K/2) x N], u32 = {k even, k odd}
// EPI 0: v = acc * extra[col]                      (cos-resonance scale)
// EPI 1: v = acc + bias[col] + extra[row*N+col]    (bias + residual)
// EPI 2: v = gelu(acc + bias[col])                 (FF up-proj)
// OUTM 0: store f32 C | 1: store f32 C and bf16 Cb | 2: store bf16 Cb only
//
// 128x128 block, BK=32, 8 waves, wave = 32x64 = 2x4 WMMA tiles.
// LDS fragment-major: sX[buf][(tile*32+lane)*8 dwords] = lane's 16 halfwords.
// Double buffered; async staging overlapped with WMMA, 1 barrier per step.
// ---------------------------------------------------------------------------
#define BM 128
#define BN 128
#define BK 32

template <int EPI, int OUTM>
__global__ __launch_bounds__(256)
void gemm_bf16_wmma(const unsigned short* __restrict__ A,
                    const unsigned* __restrict__ Bp,
                    float* __restrict__ C, unsigned short* __restrict__ Cb,
                    int M, int N, int K,
                    const float* __restrict__ bias, const float* __restrict__ extra) {
    __shared__ __align__(32) unsigned sA[2][8 * 32 * 8];   // 8 row-tiles, 8KB each buf
    __shared__ __align__(32) unsigned sB[2][8 * 32 * 8];   // 8 col-tiles

    const int t    = threadIdx.x;
    const int lane = t & 31;
    const int wave = t >> 5;
    const int wr   = wave & 3;     // 4 wave-rows  x 32 rows
    const int wc   = wave >> 2;    // 2 wave-cols  x 64 cols
    const int blockRow = blockIdx.y * BM;
    const int blockCol = blockIdx.x * BN;
    const int lrow  = lane & 15;
    const int lhalf = lane >> 4;

    AccU acc[2][4];
#pragma unroll
    for (int r = 0; r < 2; ++r)
#pragma unroll
        for (int c = 0; c < 4; ++c)
#pragma unroll
            for (int i = 0; i < 8; ++i) acc[r][c].e[i] = 0.0f;

    // Per-thread staging coordinates (fixed over the K loop).
    // A: idx = t + i*256 -> r = idx>>4 (0..127), kp = idx&15 (k = 2*kp)
    // B: idx = t + i*256 -> kp = idx>>7 (k = 2*kp), col = idx&127
#if !HAVE_ASYNC_LDS
    unsigned aR[8], bR[8];
#endif

    auto stage = [&](int k0, int buf) {
#pragma unroll
        for (int i = 0; i < 8; ++i) {                  // A tile
            int idx = t + i * 256;
            int r = idx >> 4, k = (idx & 15) * 2;
            int tile = r >> 4;
            int ln   = (r & 15) + (((k >> 3) & 1) << 4);      // A 16x32 stripe
            int dw   = ((k & 7) >> 1) + ((k >> 4) << 2);
            const unsigned* gp = (const unsigned*)&A[(size_t)(blockRow + r) * K + k0 + k];
            unsigned* lp = &sA[buf][(tile * 32 + ln) * 8 + dw];
#if HAVE_ASYNC_LDS
            __builtin_amdgcn_global_load_async_to_lds_b32(
                (gint_t*)gp, (lint_t*)lp, 0, 0);
#else
            aR[i] = *gp; (void)lp;
#endif
        }
#pragma unroll
        for (int i = 0; i < 8; ++i) {                  // B tile
            int idx = t + i * 256;
            int k = (idx >> 7) * 2, col = idx & 127;
            int tile = col >> 4;
            int ln   = (col & 15) + (((k >> 3) & 1) << 4);    // B 32x16 stripe
            int dw   = ((k & 7) >> 1) + ((k >> 4) << 2);
            const unsigned* gp = &Bp[(size_t)((k0 + k) >> 1) * N + blockCol + col];
            unsigned* lp = &sB[buf][(tile * 32 + ln) * 8 + dw];
#if HAVE_ASYNC_LDS
            __builtin_amdgcn_global_load_async_to_lds_b32(
                (gint_t*)gp, (lint_t*)lp, 0, 0);
#else
            bR[i] = *gp; (void)lp;
#endif
        }
    };

#if !HAVE_ASYNC_LDS
    auto stageCommit = [&](int buf) {
#pragma unroll
        for (int i = 0; i < 8; ++i) {
            int idx = t + i * 256;
            int r = idx >> 4, k = (idx & 15) * 2;
            int tile = r >> 4;
            int ln   = (r & 15) + (((k >> 3) & 1) << 4);
            int dw   = ((k & 7) >> 1) + ((k >> 4) << 2);
            sA[buf][(tile * 32 + ln) * 8 + dw] = aR[i];
        }
#pragma unroll
        for (int i = 0; i < 8; ++i) {
            int idx = t + i * 256;
            int k = (idx >> 7) * 2, col = idx & 127;
            int tile = col >> 4;
            int ln   = (col & 15) + (((k >> 3) & 1) << 4);
            int dw   = ((k & 7) >> 1) + ((k >> 4) << 2);
            sB[buf][(tile * 32 + ln) * 8 + dw] = bR[i];
        }
    };
#endif

    const int nk = K / BK;
    stage(0, 0);
#if HAVE_ASYNC_LDS
    __builtin_amdgcn_s_wait_asynccnt(0);
#else
    stageCommit(0);
#endif
    __syncthreads();

    const int tA = wr * 2;   // this wave's row tiles: tA, tA+1
    const int tB = wc * 4;   // this wave's col tiles: tB .. tB+3

    for (int kt = 0; kt < nk; ++kt) {
        int cur = kt & 1;
        if (kt + 1 < nk) stage((kt + 1) * BK, cur ^ 1);

        v16bf afv[2], bfv[4];
#pragma unroll
        for (int rt = 0; rt < 2; ++rt)
            afv[rt] = *(const v16bf*)&sA[cur][((tA + rt) * 32 + lane) * 8];
#pragma unroll
        for (int ct = 0; ct < 4; ++ct)
            bfv[ct] = *(const v16bf*)&sB[cur][((tB + ct) * 32 + lane) * 8];

#pragma unroll
        for (int rt = 0; rt < 2; ++rt)
#pragma unroll
            for (int ct = 0; ct < 4; ++ct)
                acc[rt][ct].v = __builtin_amdgcn_wmma_f32_16x16x32_bf16(
                    false, afv[rt], false, bfv[ct],
                    (short)0, acc[rt][ct].v, false, false);

        if (kt + 1 < nk) {
#if HAVE_ASYNC_LDS
            __builtin_amdgcn_s_wait_asynccnt(0);
#else
            stageCommit(cur ^ 1);
#endif
        }
        __syncthreads();
    }

    // ---- fused epilogue + store (C/D layout: vgpr i -> row, lane -> col) ----
#pragma unroll
    for (int rt = 0; rt < 2; ++rt) {
#pragma unroll
        for (int ct = 0; ct < 4; ++ct) {
            int col = blockCol + wc * 64 + ct * 16 + lrow;
#pragma unroll
            for (int i = 0; i < 8; ++i) {
                int row = blockRow + wr * 32 + rt * 16 + (lhalf << 3) + i;
                float v = acc[rt][ct].e[i];
                if (EPI == 0) {
                    v *= extra[col];
                } else if (EPI == 1) {
                    v += bias[col] + extra[(size_t)row * N + col];
                } else {
                    v = gelu_exact(v + bias[col]);
                }
                if (OUTM != 2) C[(size_t)row * N + col] = v;
                if (OUTM != 0) Cb[(size_t)row * N + col] = bf16h(v);
            }
        }
    }
}

// ---------------------------------------------------------------------------
// weight pack: W (K x N f32, row-major) -> u32[(K/2) x N], {bf16 k, bf16 k+1}
// ---------------------------------------------------------------------------
__global__ __launch_bounds__(256)
void pack_w_kernel(const float* __restrict__ W, unsigned* __restrict__ out, int N) {
    int col = blockIdx.x * 256 + threadIdx.x;
    int kp  = blockIdx.y;
    unsigned lo = bf16bits(W[(size_t)(2 * kp) * N + col]);
    unsigned hi = bf16bits(W[(size_t)(2 * kp + 1) * N + col]);
    out[(size_t)kp * N + col] = lo | (hi << 16);
}

// f32 -> bf16 elementwise
__global__ void f32_to_bf16_kernel(const float* __restrict__ src,
                                   unsigned short* __restrict__ dst, int n) {
    int i = blockIdx.x * 256 + threadIdx.x;
    if (i < n) dst[i] = bf16h(src[i]);
}

// ---------------------------------------------------------------------------
// cos(pi * resonance_freqs) -> per-column scale vector (768)
// ---------------------------------------------------------------------------
__global__ void cosrf_kernel(const float* __restrict__ rf, float* __restrict__ out) {
    int i = blockIdx.x * 256 + threadIdx.x;
    if (i < 768) out[i] = cosf(rf[i] * 3.14159265358979323846f);
}

// ---------------------------------------------------------------------------
// summary[b, col] = mean_s heads[b, s, col]          (col = h*96+d)
// ---------------------------------------------------------------------------
__global__ __launch_bounds__(256)
void colmean_kernel(const float* __restrict__ heads, float* __restrict__ summary, int S) {
    int col = blockIdx.x * 256 + threadIdx.x;  // 0..767
    int b   = blockIdx.y;
    float s = 0.0f;
    for (int i = 0; i < S; ++i) s += heads[(size_t)(b * S + i) * 768 + col];
    summary[b * 768 + col] = s * (1.0f / (float)S);
}

// ---------------------------------------------------------------------------
// polarity -> impedance -> coef  (per batch; tiny network)
// ---------------------------------------------------------------------------
__global__ __launch_bounds__(256)
void polimp_kernel(const float* __restrict__ summary, const float* __restrict__ pW,
                   const float* __restrict__ pb, const float* __restrict__ iw1,
                   const float* __restrict__ ib1, const float* __restrict__ iw2,
                   const float* __restrict__ ib2, float* __restrict__ coef) {
    int b = blockIdx.x;
    int t = threadIdx.x;
    int h = t >> 5, p = t & 31;
    __shared__ float pol[8][32];

    float a = 0.0f;
    for (int d = 0; d < 96; ++d)
        a += summary[b * 768 + h * 96 + d] * pW[(h * 96 + d) * 32 + p];
    a = tanhf(a + pb[h * 32 + p]);
    pol[h][p] = a;
    __syncthreads();

    float s2 = 0.0f;
    for (int q = 0; q < 32; ++q) { float v = pol[h][q]; s2 += v * v; }
    float inv = 1.0f / fmaxf(sqrtf(s2), 1e-12f);
    __syncthreads();
    pol[h][p] = a * inv;
    __syncthreads();

    if (t < 64) {
        int i = t >> 3, j = t & 7;
        float dot = 0.0f;
        for (int q = 0; q < 32; ++q) dot += pol[i][q] * pol[j][q];
        float acc = 0.0f;
        for (int q = 0; q < 16; ++q) {
            float z = dot * iw1[q] + ib1[q];
            acc += gelu_exact(z) * iw2[q];
        }
        float z  = acc + ib2[0];
        float sp = (z > 20.0f) ? z : log1pf(expf(z));
        coef[b * 64 + t] = (i == j) ? 0.0f : 1.0f / (1.0f + sp);
    }
}

// ---------------------------------------------------------------------------
// Causal cumsum over S + cross-head transfer, in place on heads (f32),
// plus a bf16 copy of the merged result for the following WMMA GEMM.
// Block = one (batch, d) pair; scans all 8 head-channels in LDS.
// ---------------------------------------------------------------------------
__global__ __launch_bounds__(256)
void scan_mix_kernel(float* __restrict__ heads, unsigned short* __restrict__ headsB,
                     const float* __restrict__ coef, int S) {
    int d = blockIdx.x;   // 0..95
    int b = blockIdx.y;
    int t = threadIdx.x;
    __shared__ float cum[8][1024];
    __shared__ float tsum[256];
    __shared__ float cf[64];
    if (t < 64) cf[t] = coef[b * 64 + t];
    __syncthreads();

    for (int j = 0; j < 8; ++j) {
        float v[4];
        float run = 0.0f;
#pragma unroll
        for (int u = 0; u < 4; ++u) {
            int s = t * 4 + u;
            v[u] = heads[(size_t)(b * S + s) * 768 + j * 96 + d];
            run += v[u];
        }
        tsum[t] = run;
        __syncthreads();
        for (int off = 1; off < 256; off <<= 1) {   // Hillis-Steele inclusive scan
            float add = (t >= off) ? tsum[t - off] : 0.0f;
            __syncthreads();
            tsum[t] += add;
            __syncthreads();
        }
        float acc = tsum[t] - run;                  // exclusive prefix
#pragma unroll
        for (int u = 0; u < 4; ++u) { acc += v[u]; cum[j][t * 4 + u] = acc; }
        __syncthreads();
    }

    const float scale = 0.1f / (float)S;
#pragma unroll
    for (int u = 0; u < 4; ++u) {
        int s = t * 4 + u;
        size_t base = (size_t)(b * S + s) * 768 + d;
#pragma unroll
        for (int h = 0; h < 8; ++h) {
            float m = 0.0f;
#pragma unroll
            for (int j = 0; j < 8; ++j) m += cf[h * 8 + j] * cum[j][s];
            float nv = heads[base + (size_t)h * 96] + scale * m;
            heads[base + (size_t)h * 96]  = nv;
            headsB[base + (size_t)h * 96] = bf16h(nv);
        }
    }
}

// ---------------------------------------------------------------------------
// Double LayerNorm: H(f32) = LN1(X2)*g1+b1 ; Yb(bf16) = LN2(H)*fg+fb
// ---------------------------------------------------------------------------
static __device__ __forceinline__ float block_sum(float v, float* red, int t) {
    red[t] = v;
    __syncthreads();
    for (int off = 128; off > 0; off >>= 1) {
        if (t < off) red[t] += red[t + off];
        __syncthreads();
    }
    float r = red[0];
    __syncthreads();
    return r;
}

__global__ __launch_bounds__(256)
void ln2_kernel(const float* __restrict__ X2, float* __restrict__ H,
                unsigned short* __restrict__ Yb,
                const float* __restrict__ g1, const float* __restrict__ b1,
                const float* __restrict__ fg, const float* __restrict__ fb) {
    int row = blockIdx.x, t = threadIdx.x;
    __shared__ float red[256];
    const size_t base = (size_t)row * 768;
    float x0 = X2[base + t], x1 = X2[base + t + 256], x2 = X2[base + t + 512];

    float m = block_sum(x0 + x1 + x2, red, t) * (1.0f / 768.0f);
    float d0 = x0 - m, d1 = x1 - m, d2 = x2 - m;
    float v = block_sum(d0 * d0 + d1 * d1 + d2 * d2, red, t) * (1.0f / 768.0f);
    float r = rsqrtf(v + 1e-5f);
    float h0 = d0 * r * g1[t]       + b1[t];
    float h1 = d1 * r * g1[t + 256] + b1[t + 256];
    float h2 = d2 * r * g1[t + 512] + b1[t + 512];
    H[base + t] = h0; H[base + t + 256] = h1; H[base + t + 512] = h2;

    float m2 = block_sum(h0 + h1 + h2, red, t) * (1.0f / 768.0f);
    float e0 = h0 - m2, e1 = h1 - m2, e2 = h2 - m2;
    float v2 = block_sum(e0 * e0 + e1 * e1 + e2 * e2, red, t) * (1.0f / 768.0f);
    float r2 = rsqrtf(v2 + 1e-5f);
    Yb[base + t]       = bf16h(e0 * r2 * fg[t]       + fb[t]);
    Yb[base + t + 256] = bf16h(e1 * r2 * fg[t + 256] + fb[t + 256]);
    Yb[base + t + 512] = bf16h(e2 * r2 * fg[t + 512] + fb[t + 512]);
}

// ---------------------------------------------------------------------------
// host-side orchestration (graph-capture safe)
// ---------------------------------------------------------------------------
extern "C" void kernel_launch(void* const* d_in, const int* in_sizes, int n_in,
                              void* d_out, int out_size, void* d_ws, size_t ws_size,
                              hipStream_t stream) {
    (void)in_sizes; (void)n_in; (void)out_size; (void)ws_size;
    const float* x    = (const float*)d_in[0];
    const float* Wtri = (const float*)d_in[1];
    const float* rf   = (const float*)d_in[2];
    const float* pW   = (const float*)d_in[3];
    const float* pb   = (const float*)d_in[4];
    const float* iw1  = (const float*)d_in[5];
    const float* ib1  = (const float*)d_in[6];
    const float* iw2  = (const float*)d_in[7];
    const float* ib2  = (const float*)d_in[8];
    const float* oW   = (const float*)d_in[9];
    const float* ob   = (const float*)d_in[10];
    const float* g1   = (const float*)d_in[11];
    const float* b1   = (const float*)d_in[12];
    const float* fg   = (const float*)d_in[13];
    const float* fb   = (const float*)d_in[14];
    const float* fW1  = (const float*)d_in[15];
    const float* fb1  = (const float*)d_in[16];
    const float* fW2  = (const float*)d_in[17];
    const float* fb2  = (const float*)d_in[18];

    const int Bb = 4, Ss = 1024, Dd = 768, Ff = 3072, Ll = 12;
    const int Mr = Bb * Ss;                       // 4096 rows

    char* wsb = (char*)d_ws;
    size_t o = 0;
    auto allocF = [&](size_t n) { float* p = (float*)(wsb + o); o += n * 4; return p; };
    auto allocH = [&](size_t n) { unsigned short* p = (unsigned short*)(wsb + o); o += n * 2; return p; };
    auto allocU = [&](size_t n) { unsigned* p = (unsigned*)(wsb + o); o += n * 4; return p; };

    float*          bufX   = allocF((size_t)Mr * Dd);   // activations f32 (residual)
    float*          bufX2  = allocF((size_t)Mr * Dd);   // attn-out + residual
    float*          bufHd  = allocF((size_t)Mr * Dd);   // trigram heads f32
    float*          bufH   = allocF((size_t)Mr * Dd);   // post-LN1 h f32 (residual)
    unsigned short* bufXb  = allocH((size_t)Mr * Dd);   // bf16 of bufX   (GEMM1 A)
    unsigned short* bufHdB = allocH((size_t)Mr * Dd);   // bf16 merged    (GEMM2 A)
    unsigned short* bufYb  = allocH((size_t)Mr * Dd);   // bf16 LN2 out   (GEMM3 A)
    unsigned short* bufGb  = allocH((size_t)Mr * Ff);   // bf16 FF hidden (GEMM4 A)
    unsigned*       wTri   = allocU((size_t)(Dd / 2) * Dd);  // packed weights
    unsigned*       wOut   = allocU((size_t)(Dd / 2) * Dd);
    unsigned*       wFF1   = allocU((size_t)(Dd / 2) * Ff);
    unsigned*       wFF2   = allocU((size_t)(Ff / 2) * Dd);
    float*          summary = allocF((size_t)Bb * Dd);
    float*          coef    = allocF((size_t)Bb * 64);
    float*          cosrf   = allocF(768);

    (void)hipMemcpyAsync(bufX, x, sizeof(float) * (size_t)Mr * Dd,
                         hipMemcpyDeviceToDevice, stream);
    f32_to_bf16_kernel<<<(Mr * Dd + 255) / 256, 256, 0, stream>>>(x, bufXb, Mr * Dd);

    const dim3 blk(256);
    const dim3 gD(Dd / BN, Mr / BM);   // 6 x 32
    const dim3 gF(Ff / BN, Mr / BM);   // 24 x 32

    for (int l = 0; l < Ll; ++l) {
        cosrf_kernel<<<3, blk, 0, stream>>>(rf + (size_t)l * 768, cosrf);

        // heads = (h @ W_tri) * cos(pi*rf)
        pack_w_kernel<<<dim3(Dd / 256, Dd / 2), blk, 0, stream>>>(
            Wtri + (size_t)l * Dd * Dd, wTri, Dd);
        gemm_bf16_wmma<0, 0><<<gD, blk, 0, stream>>>(
            bufXb, wTri, bufHd, nullptr, Mr, Dd, Dd, nullptr, cosrf);

        colmean_kernel<<<dim3(3, Bb), blk, 0, stream>>>(bufHd, summary, Ss);
        polimp_kernel<<<Bb, blk, 0, stream>>>(
            summary, pW + (size_t)l * 8 * 96 * 32, pb + (size_t)l * 256,
            iw1 + l * 16, ib1 + l * 16, iw2 + l * 16, ib2 + l, coef);
        scan_mix_kernel<<<dim3(96, Bb), blk, 0, stream>>>(bufHd, bufHdB, coef, Ss);

        // out = merged @ out_W + out_b + residual
        pack_w_kernel<<<dim3(Dd / 256, Dd / 2), blk, 0, stream>>>(
            oW + (size_t)l * Dd * Dd, wOut, Dd);
        gemm_bf16_wmma<1, 0><<<gD, blk, 0, stream>>>(
            bufHdB, wOut, bufX2, nullptr, Mr, Dd, Dd, ob + (size_t)l * Dd, bufX);

        ln2_kernel<<<Mr, blk, 0, stream>>>(
            bufX2, bufH, bufYb,
            g1 + (size_t)l * Dd, b1 + (size_t)l * Dd,
            fg + (size_t)l * Dd, fb + (size_t)l * Dd);

        // G = gelu(y @ ff_W1 + ff_b1)    (bf16-only output: feeds GEMM4 A)
        pack_w_kernel<<<dim3(Ff / 256, Dd / 2), blk, 0, stream>>>(
            fW1 + (size_t)l * Dd * Ff, wFF1, Ff);
        gemm_bf16_wmma<2, 2><<<gF, blk, 0, stream>>>(
            bufYb, wFF1, nullptr, bufGb, Mr, Ff, Dd, fb1 + (size_t)l * Ff, nullptr);

        // next_x = G @ ff_W2 + ff_b2 + h   (f32 out + bf16 copy for next GEMM1)
        pack_w_kernel<<<dim3(Dd / 256, Ff / 2), blk, 0, stream>>>(
            fW2 + (size_t)l * Ff * Dd, wFF2, Dd);
        float* dst = (l == Ll - 1) ? (float*)d_out : bufX;
        gemm_bf16_wmma<1, 1><<<gD, blk, 0, stream>>>(
            bufGb, wFF2, dst, bufXb, Mr, Dd, Ff, fb2 + (size_t)l * Dd, bufH);
    }
}
